// AllenNLPLSTMLayer_69114613729975
// MI455X (gfx1250) — matrix operand order repaired and checked by reading, not compiled, not run on hardware
//
#include <hip/hip_runtime.h>
#include <hip/hip_bf16.h>

// ---------------------------------------------------------------------------
// Stacked BiLSTM (AllenNLP gate order i,f,g,o) for MI455X / gfx1250.
//  * Input projections: bf16 WMMA GEMM, LDS-staged A slab (ds_load_b128),
//    2x-unrolled ping-pong K loop (register double buffering with NO copies).
//  * Recurrence: one small WMMA kernel per timestep, chained on the stream;
//    h state double-buffered in bf16, shared per-block via LDS.
// Padded dims: H 500->512 (HP), gates 2000->2048 (GP). Zero-padded hidden
// units stay exactly 0 forever (g=tanh(0)=0 => c_new=0), so padding is inert.
// ---------------------------------------------------------------------------

typedef __bf16 bf16_t;
typedef __attribute__((ext_vector_type(16))) __bf16 v16bf;
typedef __attribute__((ext_vector_type(8)))  __bf16 v8bf;
typedef __attribute__((ext_vector_type(8)))  float  v8f;

#define B_  64
#define T_  512
#define H_  500
#define HP  512
#define GP  2048   // 4*HP

#define WMMA_BF16(a, b, c) \
  __builtin_amdgcn_wmma_f32_16x16x32_bf16(false, (a), false, (b), (short)0, (c), false, false)

// ---- WMMA fragment loaders (wave32; layouts per CDNA5 ISA 7.12.2) ----------
// A (16x32 bf16, rows striped in-lane): lane = 16*half + m.
//   elems 0..7  -> K = half*8 + e ;  elems 8..15 -> K = 16 + half*8 + (e-8)
__device__ __forceinline__ v16bf load_a_frag(const bf16_t* p0, int lda) {
  const int lane = threadIdx.x & 31;
  const int m = lane & 15, half = lane >> 4;
  const bf16_t* p = p0 + (size_t)m * lda + half * 8;
  union { v16bf v; v8bf h[2]; } u;
  u.h[0] = *(const v8bf*)(p);        // K = half*8 + 0..7
  u.h[1] = *(const v8bf*)(p + 16);   // K = 16 + half*8 + 0..7
  return u.v;
}

// B (32x16 bf16, K striped across elements): lane = 16*half + n,
//   elems 0..7 -> K = half*8 + e ; elems 8..15 -> K = 16 + half*8 + (e-8). We read B[k][n] = W[n*ldw + k] (W row-major).
__device__ __forceinline__ v16bf load_b_frag(const bf16_t* p0, int ldw) {
  const int lane = threadIdx.x & 31;
  const int n = lane & 15, half = lane >> 4;
  const bf16_t* p = p0 + (size_t)n * ldw + half * 8;
  union { v16bf v; v8bf h[2]; } u;
  u.h[0] = *(const v8bf*)(p);
  u.h[1] = *(const v8bf*)(p + 16);
  return u.v;
}

// ---- Big input-projection GEMM: out[M,GP] = A[M,Kp] * W[GP,Kp]^T + bias ----
// Block = 8 waves; block covers 16 rows x 512 cols; each wave a 16x64 strip.
// A slab (16 x Kp) staged in LDS, row stride padded +8 bf16 -> conflict-free.
// K loop: 2x unrolled ping-pong over register sets {a0,b0} / {a1,b1}.
__global__ void __launch_bounds__(256)
k_gemm_xgates(const bf16_t* __restrict__ A, const bf16_t* __restrict__ W,
              const float* __restrict__ bias, float* __restrict__ out, int Kp) {
  __shared__ bf16_t As[16 * (1024 + 8)];
  const int ldsA = Kp + 8;
  const int m0 = blockIdx.y * 16;

  // Cooperative stage of the A slab (16 x Kp bf16) into LDS.
  const int cpr = Kp >> 3;                     // 16B chunks per row
  for (int idx = threadIdx.x; idx < 16 * cpr; idx += 256) {
    const int r = idx / cpr;
    const int cc = (idx - r * cpr) << 3;
    *(v8bf*)(&As[r * ldsA + cc]) = *(const v8bf*)(A + (size_t)(m0 + r) * Kp + cc);
  }
  __syncthreads();

  const int wave = threadIdx.x >> 5;
  const int n0 = blockIdx.x * 512 + wave * 64;
  const bf16_t* wBase = W + (size_t)n0 * Kp;

  v8f acc[4] = {{}, {}, {}, {}};
  v16bf a0, a1, b0[4], b1[4];
  a0 = load_a_frag(As, ldsA);
#pragma unroll
  for (int q = 0; q < 4; ++q) b0[q] = load_b_frag(wBase + (size_t)q * 16 * Kp, Kp);

  for (int k0 = 0; k0 < Kp; k0 += 64) {        // Kp is a multiple of 64
    // stage A: issue loads for k0+32, compute slab k0 from set 0
    a1 = load_a_frag(As + k0 + 32, ldsA);
#pragma unroll
    for (int q = 0; q < 4; ++q)
      b1[q] = load_b_frag(wBase + (size_t)q * 16 * Kp + k0 + 32, Kp);
#pragma unroll
    for (int q = 0; q < 4; ++q) acc[q] = WMMA_BF16(a0, b0[q], acc[q]);
    // stage B: issue loads for k0+64, compute slab k0+32 from set 1
    if (k0 + 64 < Kp) {
      a0 = load_a_frag(As + k0 + 64, ldsA);
#pragma unroll
      for (int q = 0; q < 4; ++q)
        b0[q] = load_b_frag(wBase + (size_t)q * 16 * Kp + k0 + 64, Kp);
      __builtin_prefetch(wBase + k0 + 96, 0, 1); // gfx1250 global_prefetch_b8
    }
#pragma unroll
    for (int q = 0; q < 4; ++q) acc[q] = WMMA_BF16(a1, b1[q], acc[q]);
  }

  const int lane = threadIdx.x & 31;
  const int n = lane & 15, half = lane >> 4;
#pragma unroll
  for (int q = 0; q < 4; ++q) {
    const int col = n0 + q * 16 + n;
    const float bv = bias[col];
#pragma unroll
    for (int v = 0; v < 8; ++v) {
      const int m = m0 + v + 8 * half;
      out[(size_t)m * GP + col] = acc[q][v] + bv;
    }
  }
}

// ---- One LSTM timestep: gates = xg[:,tt,:] + h @ Whh^T ; update h,c --------
// Block = 4 waves sharing one 16-batch slab of h (LDS-staged); each wave owns
// one 16-hidden tile and accumulates all four gates (A reused 4x per slab).
// hbf is double-buffered across launches (all blocks read all columns).
__global__ void __launch_bounds__(128)
k_lstm_step(const float* __restrict__ xg, const bf16_t* __restrict__ whh,
            float* __restrict__ h, float* __restrict__ c,
            const bf16_t* __restrict__ hbf_in, bf16_t* __restrict__ hbf_out,
            const int* __restrict__ lens, int tt,
            float* __restrict__ y, int ycol) {
  __shared__ bf16_t Hs[16 * (HP + 8)];
  const int ldsH = HP + 8;
  const int m0 = blockIdx.y * 16;

  const int cpr = HP >> 3;
  for (int idx = threadIdx.x; idx < 16 * cpr; idx += 128) {
    const int r = idx / cpr;
    const int cc = (idx - r * cpr) << 3;
    *(v8bf*)(&Hs[r * ldsH + cc]) = *(const v8bf*)(hbf_in + (size_t)(m0 + r) * HP + cc);
  }
  __syncthreads();

  const int wave = threadIdx.x >> 5;
  const int n0 = (blockIdx.x * 4 + wave) * 16;   // hidden tile (0..511)
  const bf16_t* wBase = whh + (size_t)n0 * HP;   // gate q at +q*HP*HP

  v8f acc[4] = {{}, {}, {}, {}};
  v16bf a0, a1, b0[4], b1[4];
  a0 = load_a_frag(Hs, ldsH);
#pragma unroll
  for (int q = 0; q < 4; ++q) b0[q] = load_b_frag(wBase + (size_t)q * HP * HP, HP);

  for (int k0 = 0; k0 < HP; k0 += 64) {          // HP = 512, multiple of 64
    a1 = load_a_frag(Hs + k0 + 32, ldsH);
#pragma unroll
    for (int q = 0; q < 4; ++q)
      b1[q] = load_b_frag(wBase + (size_t)q * HP * HP + k0 + 32, HP);
#pragma unroll
    for (int q = 0; q < 4; ++q) acc[q] = WMMA_BF16(a0, b0[q], acc[q]);
    if (k0 + 64 < HP) {
      a0 = load_a_frag(Hs + k0 + 64, ldsH);
#pragma unroll
      for (int q = 0; q < 4; ++q)
        b0[q] = load_b_frag(wBase + (size_t)q * HP * HP + k0 + 64, HP);
    }
#pragma unroll
    for (int q = 0; q < 4; ++q) acc[q] = WMMA_BF16(a1, b1[q], acc[q]);
  }

  const int lane = threadIdx.x & 31;
  const int n = lane & 15, half = lane >> 4;
  const int j = n0 + n;
#pragma unroll
  for (int v = 0; v < 8; ++v) {
    const int bb = m0 + v + 8 * half;           // batch index
    const float* g = xg + ((size_t)bb * T_ + tt) * GP;
    const float gi = acc[0][v] + g[j];
    const float gf = acc[1][v] + g[HP + j];
    const float gg = acc[2][v] + g[2 * HP + j];
    const float go = acc[3][v] + g[3 * HP + j];
    const float ig = 1.f / (1.f + __expf(-gi));
    const float fg = 1.f / (1.f + __expf(-gf));
    const float cg = tanhf(gg);
    const float og = 1.f / (1.f + __expf(-go));
    const size_t s = (size_t)bb * HP + j;
    const float c_old = c[s];
    const float h_old = h[s];
    const float c_new = fg * c_old + ig * cg;
    const float h_new = og * tanhf(c_new);
    const bool mk = tt < lens[bb];
    const float ho = mk ? h_new : h_old;
    const float co = mk ? c_new : c_old;
    h[s] = ho;
    c[s] = co;
    hbf_out[s] = (bf16_t)ho;
    if (j < H_)
      y[((size_t)bb * T_ + tt) * (2 * H_) + ycol + j] = mk ? ho : 0.f;
  }
}

// ---- Packing / init helpers ------------------------------------------------
// W [4*H_, Kin] f32 -> out [GP, Kp] bf16, zero-padded per gate block.
__global__ void k_pack_weight(const float* __restrict__ W, bf16_t* __restrict__ out,
                              int Kin, int Kp) {
  const size_t idx = (size_t)blockIdx.x * blockDim.x + threadIdx.x;
  const size_t total = (size_t)GP * Kp;
  if (idx >= total) return;
  const int k = (int)(idx % Kp);
  const int r = (int)(idx / Kp);
  const int q = r >> 9;            // / HP
  const int j = r & (HP - 1);
  float v = 0.f;
  if (j < H_ && k < Kin) v = W[(size_t)(q * H_ + j) * Kin + k];
  out[idx] = (bf16_t)v;
}

__global__ void k_pack_bias(const float* __restrict__ b, float* __restrict__ out) {
  const int r = blockIdx.x * blockDim.x + threadIdx.x;
  if (r >= GP) return;
  const int q = r >> 9, j = r & (HP - 1);
  out[r] = (j < H_) ? b[q * H_ + j] : 0.f;
}

// in [B*T, Kin] f32 -> out [B*T, Kp] bf16 (zero-pad columns Kin..Kp)
__global__ void k_pack_input(const float* __restrict__ in, bf16_t* __restrict__ out,
                             int Kin, int Kp) {
  const size_t idx = (size_t)blockIdx.x * blockDim.x + threadIdx.x;
  const size_t total = (size_t)(B_ * T_) * Kp;
  if (idx >= total) return;
  const size_t r = idx / Kp;
  const int k = (int)(idx % Kp);
  out[idx] = (bf16_t)((k < Kin) ? in[r * Kin + k] : 0.f);
}

__global__ void k_zero_state(float* h, float* c, bf16_t* a, bf16_t* b2) {
  const int i = blockIdx.x * blockDim.x + threadIdx.x;
  if (i < B_ * HP) {
    h[i] = 0.f; c[i] = 0.f;
    a[i] = (bf16_t)0.f; b2[i] = (bf16_t)0.f;
  }
}

// ---------------------------------------------------------------------------
extern "C" void kernel_launch(void* const* d_in, const int* in_sizes, int n_in,
                              void* d_out, int out_size, void* d_ws, size_t ws_size,
                              hipStream_t stream) {
  (void)in_sizes; (void)n_in; (void)out_size; (void)ws_size;
  const float* seqs = (const float*)d_in[0];
  const int*   lens = (const int*)d_in[1];   // lengths (cast per harness doc)
  const float* Wih[2][2] = {{(const float*)d_in[2],  (const float*)d_in[5]},
                            {(const float*)d_in[8],  (const float*)d_in[11]}};
  const float* Whh[2][2] = {{(const float*)d_in[3],  (const float*)d_in[6]},
                            {(const float*)d_in[9],  (const float*)d_in[12]}};
  const float* Bv[2][2]  = {{(const float*)d_in[4],  (const float*)d_in[7]},
                            {(const float*)d_in[10], (const float*)d_in[13]}};

  // Workspace carve (~474 MB total).
  char* base = (char*)d_ws;
  size_t off = 0;
  auto carve = [&](size_t bytes) -> char* {
    char* p = base + off;
    off += (bytes + 255) & ~(size_t)255;
    return p;
  };
  float*  xg    = (float*) carve((size_t)B_ * T_ * GP * 4);     // 256 MB gate preacts
  bf16_t* xbf   = (bf16_t*)carve((size_t)B_ * T_ * 1024 * 2);   //  64 MB bf16 layer input
  float*  y0    = (float*) carve((size_t)B_ * T_ * 1000 * 4);   // 128 MB layer-0 concat out
  bf16_t* wih   = (bf16_t*)carve((size_t)GP * 1024 * 2);        //   4 MB
  bf16_t* whhp  = (bf16_t*)carve((size_t)GP * HP * 2);          //   2 MB
  float*  biasp = (float*) carve((size_t)GP * 4);
  float*  hst   = (float*) carve((size_t)B_ * HP * 4);
  float*  cst   = (float*) carve((size_t)B_ * HP * 4);
  bf16_t* hbfA  = (bf16_t*)carve((size_t)B_ * HP * 2);
  bf16_t* hbfB  = (bf16_t*)carve((size_t)B_ * HP * 2);

  for (int l = 0; l < 2; ++l) {
    const int Kin = l ? 2 * H_ : 512;
    const int Kp  = l ? 1024 : 512;
    {
      const size_t total = (size_t)(B_ * T_) * Kp;
      k_pack_input<<<(unsigned)((total + 255) / 256), 256, 0, stream>>>(
          l ? (const float*)y0 : seqs, xbf, Kin, Kp);
    }
    for (int d = 0; d < 2; ++d) {
      {
        const size_t tw = (size_t)GP * Kp;
        k_pack_weight<<<(unsigned)((tw + 255) / 256), 256, 0, stream>>>(
            Wih[l][d], wih, Kin, Kp);
        const size_t th = (size_t)GP * HP;
        k_pack_weight<<<(unsigned)((th + 255) / 256), 256, 0, stream>>>(
            Whh[l][d], whhp, H_, HP);
        k_pack_bias<<<(GP + 255) / 256, 256, 0, stream>>>(Bv[l][d], biasp);
      }
      // xg[b*T+t, :] = x @ Wih^T + b   (M=32768, N=2048, K=Kp)
      k_gemm_xgates<<<dim3(GP / 512, (B_ * T_) / 16), 256, 0, stream>>>(
          xbf, wih, biasp, xg, Kp);
      k_zero_state<<<(B_ * HP + 255) / 256, 256, 0, stream>>>(hst, cst, hbfA, hbfB);

      float* yout = l ? (float*)d_out : y0;
      const int ycol = d ? H_ : 0;
      bf16_t* cur = hbfA; bf16_t* nxt = hbfB;
      for (int t = 0; t < T_; ++t) {
        const int tt = d ? (T_ - 1 - t) : t;   // bwd direction scans reversed
        k_lstm_step<<<dim3(HP / 64, B_ / 16), 128, 0, stream>>>(
            xg, whhp, hst, cst, cur, nxt, lens, tt, yout, ycol);
        bf16_t* tmp = cur; cur = nxt; nxt = tmp;
      }
    }
  }
}
